// MultiHeadAttention_31937376813246
// MI455X (gfx1250) — matrix operand order repaired and checked by reading, not compiled, not run on hardware
//
#include <hip/hip_runtime.h>

// ---------------------------------------------------------------------------
// MHA forward for MI455X (gfx1250, wave32, WMMA).
//   B=2, T=2048, C=1024, H=16, D=64.  ~70 GFLOP, compute-bound -> all matmuls
//   run through v_wmma_f32_16x16x32_f16 (f16 inputs, f32 accumulate).
//   Global latency hidden via register software-pipelining (GEMMs) and
//   async-to-LDS double buffering (flash attention K tiles).
// ---------------------------------------------------------------------------

typedef __attribute__((ext_vector_type(16))) _Float16 v16h;
typedef __attribute__((ext_vector_type(8)))  _Float16 v8h;
typedef __attribute__((ext_vector_type(4)))  _Float16 v4h;
typedef __attribute__((ext_vector_type(8)))  float    v8f;

union Frag {            // one WMMA 16-bit operand: 8 VGPRs, loaded as 2x b128
    v16h v;
    v8h  h[2];
};

#ifndef __has_builtin
#define __has_builtin(x) 0
#endif

#if __has_builtin(__builtin_amdgcn_global_load_async_to_lds_b128)
#define HAVE_ASYNC_LDS 1
// builtin signature (from hipcc diagnostic): (v4i as1*, v4i as3*, Ii, Ii)
typedef int vint4 __attribute__((__vector_size__(16)));
#define GAS __attribute__((address_space(1)))
#define LAS __attribute__((address_space(3)))
__device__ __forceinline__ void async_copy16(const _Float16* g, _Float16* l) {
    // fire-and-forget 16B/lane copy, landed by the async engine (ASYNCcnt)
    __builtin_amdgcn_global_load_async_to_lds_b128(
        (GAS vint4*)(void*)const_cast<_Float16*>(g),
        (LAS vint4*)(void*)l, 0, 0);
}
__device__ __forceinline__ void wait_async0() {
#if __has_builtin(__builtin_amdgcn_s_wait_asynccnt)
    __builtin_amdgcn_s_wait_asynccnt(0);
#else
    asm volatile("s_wait_asynccnt 0x0" ::: "memory");
#endif
}
#else
#define HAVE_ASYNC_LDS 0
#endif

// bitmask swizzle within a wave32 (and_mask=0x1f, or=MASK, xor=0) -- used for
// row reductions across the 16-lane groups of the WMMA C/D layout.
template <int MASK>
__device__ __forceinline__ float swz_xor(float x) {
    return __int_as_float(
        __builtin_amdgcn_ds_swizzle(__float_as_int(x), (MASK << 5) | 0x1f));
}

#define BM 128
#define BN 128
#define BK 32
#define LDT 40   // LDS tile row stride in halfwords (32 + 8 pad -> 80B rows)

// ---------------------------------------------------------------------------
// Y = A @ W^T + bias  (W is [N,K] torch-style, so both operands stream along K)
// Software-pipelined: tile kt+1 is loaded into VGPRs while WMMAs consume tile
// kt from LDS, so VMEM latency hides under the 8 WMMAs x 8 waves per block.
// ---------------------------------------------------------------------------
template <bool A_F16, bool SPLIT_HEADS>
__global__ __launch_bounds__(256) void linear_wmma(
    const float* __restrict__ A32, const _Float16* __restrict__ A16,
    const float* __restrict__ W,   const float* __restrict__ bias,
    float* __restrict__ out32,     _Float16* __restrict__ out16,
    int M, int N, int K, float scale)
{
    __shared__ _Float16 As[BM * LDT];
    __shared__ _Float16 Bs[BN * LDT];

    const int tid   = threadIdx.x;
    const int lane  = tid & 31;
    const int wave  = tid >> 5;       // 8 waves
    const int wm    = wave >> 2;      // 0..1 : 64-row strip
    const int wn    = wave & 3;       // 0..3 : 32-col strip
    const int l16   = lane & 15;
    const int lhalf = lane >> 4;

    const int m0 = blockIdx.y * BM;
    const int n0 = blockIdx.x * BN;

    v8f acc[4][2];
#pragma unroll
    for (int mi = 0; mi < 4; ++mi)
#pragma unroll
        for (int ni = 0; ni < 2; ++ni)
#pragma unroll
            for (int r = 0; r < 8; ++r) acc[mi][ni][r] = 0.0f;

    // register staging for the in-flight global tile
    float4 aR[4];
    v8h    aR16[2];
    float4 wR[4];

    auto loadTile = [&](int k0) {
        if constexpr (A_F16) {
#pragma unroll
            for (int i = 0; i < 2; ++i) {
                int idx = tid + i * 256;          // 512 chunks of 8 halfs
                int row = idx >> 2, kc = idx & 3;
                aR16[i] = *(const v8h*)(A16 + (size_t)(m0 + row) * K + k0 + kc * 8);
            }
        } else {
#pragma unroll
            for (int i = 0; i < 4; ++i) {
                int idx = tid + i * 256;          // 1024 chunks of 4 floats
                int row = idx >> 3, kc = idx & 7;
                aR[i] = *(const float4*)(A32 + (size_t)(m0 + row) * K + k0 + kc * 4);
            }
        }
#pragma unroll
        for (int i = 0; i < 4; ++i) {
            int idx = tid + i * 256;
            int row = idx >> 3, kc = idx & 7;
            wR[i] = *(const float4*)(W + (size_t)(n0 + row) * K + k0 + kc * 4);
        }
    };

    auto stageLDS = [&]() {
        if constexpr (A_F16) {
#pragma unroll
            for (int i = 0; i < 2; ++i) {
                int idx = tid + i * 256;
                int row = idx >> 2, kc = idx & 3;
                *(v8h*)&As[row * LDT + kc * 8] = aR16[i];
            }
        } else {
#pragma unroll
            for (int i = 0; i < 4; ++i) {
                int idx = tid + i * 256;
                int row = idx >> 3, kc = idx & 7;
                v4h h = {(_Float16)aR[i].x, (_Float16)aR[i].y,
                         (_Float16)aR[i].z, (_Float16)aR[i].w};
                *(v4h*)&As[row * LDT + kc * 4] = h;
            }
        }
#pragma unroll
        for (int i = 0; i < 4; ++i) {
            int idx = tid + i * 256;
            int row = idx >> 3, kc = idx & 7;
            v4h h = {(_Float16)wR[i].x, (_Float16)wR[i].y,
                     (_Float16)wR[i].z, (_Float16)wR[i].w};
            *(v4h*)&Bs[row * LDT + kc * 4] = h;
        }
    };

    const int ntiles = K / BK;
    loadTile(0);
    for (int kt = 0; kt < ntiles; ++kt) {
        stageLDS();
        __syncthreads();
        if (kt + 1 < ntiles) loadTile((kt + 1) * BK);   // overlap with WMMAs

        // ---- fragments per the ISA VGPR layout tables ----
        Frag a[4], b[2];
#pragma unroll
        for (int mi = 0; mi < 4; ++mi) {
            const _Float16* p = &As[(wm * 64 + mi * 16 + l16) * LDT + lhalf * 8];
            a[mi].h[0] = *(const v8h*)p;        // K = c0..c0+7
            a[mi].h[1] = *(const v8h*)(p + 16); // K = 16+c0..
        }
#pragma unroll
        for (int ni = 0; ni < 2; ++ni) {
            const _Float16* p = &Bs[(wn * 32 + ni * 16 + l16) * LDT + lhalf * 8];
            b[ni].h[0] = *(const v8h*)p;        // K = c0..c0+7
            b[ni].h[1] = *(const v8h*)(p + 16); // K = 16+c0..
        }
#pragma unroll
        for (int mi = 0; mi < 4; ++mi)
#pragma unroll
            for (int ni = 0; ni < 2; ++ni)
                acc[mi][ni] = __builtin_amdgcn_wmma_f32_16x16x32_f16(
                    false, a[mi].v, false, b[ni].v, (short)0, acc[mi][ni],
                    false, false);
        __syncthreads();
    }

    // ---- epilogue: bias + scale, write out ----
#pragma unroll
    for (int mi = 0; mi < 4; ++mi)
#pragma unroll
        for (int ni = 0; ni < 2; ++ni) {
            const int col = n0 + wn * 32 + ni * 16 + l16;
            const float bv = bias[col];
#pragma unroll
            for (int r = 0; r < 8; ++r) {
                const int row = m0 + wm * 64 + mi * 16 + lhalf * 8 + r;
                const float val = (acc[mi][ni][r] + bv) * scale;
                if constexpr (SPLIT_HEADS) {
                    const int bb = row >> 11, t = row & 2047;   // T = 2048
                    const int h = col >> 6, d = col & 63;       // D = 64
                    out16[(((size_t)(bb * 16 + h) * 2048) + t) * 64 + d] =
                        (_Float16)val;
                } else {
                    out32[(size_t)row * N + col] = val;
                }
            }
        }
}

// ---------------------------------------------------------------------------
// Flash attention: one block = 128 query rows of one (b,h); 8 waves x 16 rows.
// Q frags resident in VGPRs. K tiles are copied by the async engine into a
// double-buffered LDS tile (GLOBAL_LOAD_ASYNC_TO_LDS_B128, ASYNCcnt) so the
// next block's copy overlaps the current block's 16 WMMAs; V is register-
// pipelined and transposed into LDS so the P*V B-fragment is a contiguous row.
// ---------------------------------------------------------------------------
__global__ __launch_bounds__(256) void flash_attn(
    const _Float16* __restrict__ Qh, const _Float16* __restrict__ Kh,
    const _Float16* __restrict__ Vh, _Float16* __restrict__ Oh)
{
    __shared__ _Float16 Ks[2][64 * 72];     // [key][d], double-buffered
    __shared__ _Float16 Vt[64 * 72];        // [d][key] (transposed)
    __shared__ _Float16 Ps[8 * 16 * 72];    // per-wave P transpose buffer

    const int tid   = threadIdx.x;
    const int lane  = tid & 31;
    const int wave  = tid >> 5;
    const int l16   = lane & 15;
    const int lhalf = lane >> 4;

    const int bh       = blockIdx.y;
    const size_t base  = (size_t)bh * 2048 * 64;
    const int qblock   = blockIdx.x * 128;
    const int qrow     = qblock + wave * 16;

    // per-thread staging chunk coordinates (2 chunks of 8 halfs each)
    int cr[2], cc[2];
#pragma unroll
    for (int i = 0; i < 2; ++i) {
        int idx = tid + i * 256;
        cr[i] = idx >> 3;
        cc[i] = idx & 7;
    }

    v8h vR[2];                 // in-flight V chunks (register pipeline)
#if !HAVE_ASYNC_LDS
    v8h kR[2];                 // fallback: K register pipeline
#endif

    auto issueK = [&](int kstart, int buf) {
#pragma unroll
        for (int i = 0; i < 2; ++i) {
            const _Float16* g = Kh + base + (size_t)(kstart + cr[i]) * 64 + cc[i] * 8;
#if HAVE_ASYNC_LDS
            async_copy16(g, &Ks[buf][cr[i] * 72 + cc[i] * 8]);
#else
            kR[i] = *(const v8h*)g;
            (void)buf;
#endif
        }
    };
    auto loadV = [&](int kstart) {
#pragma unroll
        for (int i = 0; i < 2; ++i)
            vR[i] = *(const v8h*)(Vh + base + (size_t)(kstart + cr[i]) * 64 + cc[i] * 8);
    };

    // Q fragments (16 rows x 64 d = 2 A-frags), straight from global
    Frag qf[2];
    {
        const _Float16* qp = Qh + base + (size_t)(qrow + l16) * 64 + lhalf * 8;
        qf[0].h[0] = *(const v8h*)(qp);
        qf[0].h[1] = *(const v8h*)(qp + 16);
        qf[1].h[0] = *(const v8h*)(qp + 32);
        qf[1].h[1] = *(const v8h*)(qp + 48);
    }

    v8f o[4];
#pragma unroll
    for (int nd = 0; nd < 4; ++nd)
#pragma unroll
        for (int r = 0; r < 8; ++r) o[nd][r] = 0.0f;
    float mrow[8], lrow[8];
#pragma unroll
    for (int r = 0; r < 8; ++r) { mrow[r] = -3.0e38f; lrow[r] = 0.0f; }

    const int nkb = qblock / 64 + 2;   // causal: key blocks covering cols <= qblock+127
    int buf = 0;
    issueK(0, 0);                      // prologue: async K(0) + V(0) in regs
    loadV(0);

    for (int kb = 0; kb < nkb; ++kb) {
        const int kstart = kb * 64;

        // ---- commit staged tile: V transpose scatter (+ K store in fallback) ----
#pragma unroll
        for (int i = 0; i < 2; ++i)
#pragma unroll
            for (int q = 0; q < 8; ++q)
                Vt[(cc[i] * 8 + q) * 72 + cr[i]] = vR[i][q];
#if HAVE_ASYNC_LDS
        wait_async0();                 // K(kb) has landed in Ks[buf]
#else
#pragma unroll
        for (int i = 0; i < 2; ++i)
            *(v8h*)&Ks[buf][cr[i] * 72 + cc[i] * 8] = kR[i];
#endif
        __syncthreads();

        // ---- issue next block's copies; they run under this block's WMMAs ----
        if (kb + 1 < nkb) {
            issueK(kstart + 64, buf ^ 1);
            loadV(kstart + 64);
            if (kb + 2 < nkb) {        // pull block kb+2 toward L2 (global_prefetch_b8)
                __builtin_prefetch(Kh + base + (size_t)(kstart + 128 + cr[0]) * 64, 0, 1);
                __builtin_prefetch(Vh + base + (size_t)(kstart + 128 + cr[0]) * 64, 0, 1);
            }
        }

        // ---- S = Q K^T  (B frag: lane=key col, contiguous d in Ks row) ----
        v8f s[4];
#pragma unroll
        for (int ni = 0; ni < 4; ++ni)
#pragma unroll
            for (int r = 0; r < 8; ++r) s[ni][r] = 0.0f;
#pragma unroll
        for (int step = 0; step < 2; ++step)
#pragma unroll
            for (int ni = 0; ni < 4; ++ni) {
                Frag bf;
                const _Float16* p = &Ks[buf][(ni * 16 + l16) * 72 + step * 32 + lhalf * 8];
                bf.h[0] = *(const v8h*)p;
                bf.h[1] = *(const v8h*)(p + 16);
                s[ni] = __builtin_amdgcn_wmma_f32_16x16x32_f16(
                    false, qf[step].v, false, bf.v, (short)0, s[ni], false, false);
            }

        // ---- causal mask (branch-free per element; wave-uniform guard) ----
        if (kstart + 63 > qrow) {
#pragma unroll
            for (int ni = 0; ni < 4; ++ni) {
                const int col = kstart + ni * 16 + l16;
#pragma unroll
                for (int r = 0; r < 8; ++r) {
                    const int row = qrow + lhalf * 8 + r;
                    s[ni][r] = (col <= row) ? s[ni][r] : -3.0e38f;
                }
            }
        }

        // ---- online softmax (reduce across 16-lane C/D groups) ----
#pragma unroll
        for (int r = 0; r < 8; ++r) {
            float cur = fmaxf(fmaxf(s[0][r], s[1][r]), fmaxf(s[2][r], s[3][r]));
            cur = fmaxf(cur, swz_xor<1>(cur));
            cur = fmaxf(cur, swz_xor<2>(cur));
            cur = fmaxf(cur, swz_xor<4>(cur));
            cur = fmaxf(cur, swz_xor<8>(cur));
            const float mnew  = fmaxf(mrow[r], cur);
            const float alpha = __expf(mrow[r] - mnew);
            mrow[r] = mnew;
            float rsum = 0.0f;
#pragma unroll
            for (int ni = 0; ni < 4; ++ni) {
                const float p = __expf(s[ni][r] - mnew);
                s[ni][r] = p;
                rsum += p;
            }
            rsum += swz_xor<1>(rsum);
            rsum += swz_xor<2>(rsum);
            rsum += swz_xor<4>(rsum);
            rsum += swz_xor<8>(rsum);
            lrow[r] = lrow[r] * alpha + rsum;
#pragma unroll
            for (int nd = 0; nd < 4; ++nd) o[nd][r] *= alpha;
        }

        // ---- P: C/D layout -> f16 A layout via per-wave LDS region ----
        _Float16* pw = &Ps[wave * 16 * 72];
#pragma unroll
        for (int ni = 0; ni < 4; ++ni)
#pragma unroll
            for (int r = 0; r < 8; ++r)
                pw[(lhalf * 8 + r) * 72 + ni * 16 + l16] = (_Float16)s[ni][r];
        __syncthreads();

        // ---- O += P V  (B frag = contiguous keys in Vt row) ----
#pragma unroll
        for (int step = 0; step < 2; ++step) {
            Frag pf;
            const _Float16* pp = &Ps[wave * 16 * 72 + l16 * 72 + step * 32 + lhalf * 8];
            pf.h[0] = *(const v8h*)pp;
            pf.h[1] = *(const v8h*)(pp + 16);
#pragma unroll
            for (int nd = 0; nd < 4; ++nd) {
                Frag vf;
                const _Float16* vp = &Vt[(nd * 16 + l16) * 72 + step * 32 + lhalf * 8];
                vf.h[0] = *(const v8h*)vp;
                vf.h[1] = *(const v8h*)(vp + 16);
                o[nd] = __builtin_amdgcn_wmma_f32_16x16x32_f16(
                    false, pf.v, false, vf.v, (short)0, o[nd], false, false);
            }
        }
        __syncthreads();               // Ks[buf]/Vt consumed; safe to restage
        buf ^= 1;
    }

    // ---- finalize: O /= l, write f16 to [B*T, C] for the projection GEMM ----
    const int b = bh >> 4, h = bh & 15;
#pragma unroll
    for (int nd = 0; nd < 4; ++nd) {
        const int col = h * 64 + nd * 16 + l16;
#pragma unroll
        for (int r = 0; r < 8; ++r) {
            const int trow = qrow + lhalf * 8 + r;
            Oh[((size_t)(b * 2048 + trow)) * 1024 + col] =
                (_Float16)(o[nd][r] / lrow[r]);
        }
    }
}

// ---------------------------------------------------------------------------
extern "C" void kernel_launch(void* const* d_in, const int* in_sizes, int n_in,
                              void* d_out, int out_size, void* d_ws, size_t ws_size,
                              hipStream_t stream) {
    const float* x  = (const float*)d_in[0];
    const float* Wq = (const float*)d_in[1];
    const float* bq = (const float*)d_in[2];
    const float* Wk = (const float*)d_in[3];
    const float* bk = (const float*)d_in[4];
    const float* Wv = (const float*)d_in[5];
    const float* bv = (const float*)d_in[6];
    const float* Wp = (const float*)d_in[7];
    const float* bp = (const float*)d_in[8];
    float* out = (float*)d_out;

    // Workspace: 4 x (4M f16 elems) = 32 MB
    const size_t NELEM = (size_t)4 * 1024 * 1024;
    _Float16* Qh = (_Float16*)d_ws;
    _Float16* Kh = Qh + NELEM;
    _Float16* Vh = Kh + NELEM;
    _Float16* Ao = Vh + NELEM;

    const dim3 blk(256);
    const dim3 gg(1024 / BN, 4096 / BM);     // (N tiles, M tiles)

    // Q/K/V projections (Q carries 1/sqrt(64) = 0.125)
    linear_wmma<false, true><<<gg, blk, 0, stream>>>(
        x, nullptr, Wq, bq, nullptr, Qh, 4096, 1024, 1024, 0.125f);
    linear_wmma<false, true><<<gg, blk, 0, stream>>>(
        x, nullptr, Wk, bk, nullptr, Kh, 4096, 1024, 1024, 1.0f);
    linear_wmma<false, true><<<gg, blk, 0, stream>>>(
        x, nullptr, Wv, bv, nullptr, Vh, 4096, 1024, 1024, 1.0f);

    // causal flash attention: grid (T/128, B*H)
    flash_attn<<<dim3(16, 32), blk, 0, stream>>>(Qh, Kh, Vh, Ao);

    // output projection -> fp32 d_out
    linear_wmma<true, false><<<gg, blk, 0, stream>>>(
        nullptr, Ao, Wp, bp, out, nullptr, 4096, 1024, 1024, 1.0f);
}
